// HybridAttention_23295902613628
// MI455X (gfx1250) — matrix operand order repaired and checked by reading, not compiled, not run on hardware
//
#include <hip/hip_runtime.h>
#include <math.h>

// ---------------------------------------------------------------------------
// Hybrid attention for MI455X (gfx1250, wave32, WMMA).
//   1) convert x_q/x_kv and (partial) weights to bf16 in workspace
//   2) bf16 WMMA GEMMs (double-buffered LDS, async global->LDS when available)
//      for Q / K_self / K_cross / V_self / V_cross
//      (K/V only need the half of the weight matrix whose heads survive;
//       V is written transposed [b][h][d][s] so PV B-fragments are contiguous)
//   3) in-place RoPE on Q (all heads) and K (all heads)
//   4) flash attention per (b,h): QK^T wmma -> online softmax -> P via LDS
//      layout swap (C-layout -> A-layout, s_wait_dscnt) -> PV wmma
//   5) final GEMM with Wo = 0.5*(Wo_self + Wo_cross), fp32 output
// ---------------------------------------------------------------------------

typedef __bf16 bf16_t;
typedef __attribute__((ext_vector_type(4)))  bf16_t bf16x4;
typedef __attribute__((ext_vector_type(8)))  bf16_t bf16x8;
typedef __attribute__((ext_vector_type(16))) bf16_t bf16x16;
typedef __attribute__((ext_vector_type(8)))  float  f32x8;
typedef __attribute__((ext_vector_type(4)))  int    v4i;

typedef __attribute__((address_space(1))) v4i* gv4i_ptr;
typedef __attribute__((address_space(3))) v4i* lv4i_ptr;

#define HIDDEN 2048
#define SEQ    2048
#define HEADS  16
#define HD     128

// ---- gfx1250 async global->LDS (ASYNCcnt) with portable fallback ----------
#if __has_builtin(__builtin_amdgcn_global_load_async_to_lds_b128)
#define HAS_ASYNC_LDS 1
#else
#define HAS_ASYNC_LDS 0
#endif

static __device__ __forceinline__ void copy16_to_lds(const bf16_t* g, bf16_t* l) {
#if HAS_ASYNC_LDS
  __builtin_amdgcn_global_load_async_to_lds_b128((gv4i_ptr)g, (lv4i_ptr)l, 0, 0);
#else
  *(bf16x8*)l = *(const bf16x8*)g;
#endif
}

template <int N>
static __device__ __forceinline__ void async_wait() {
#if HAS_ASYNC_LDS
#if __has_builtin(__builtin_amdgcn_s_wait_asynccnt)
  __builtin_amdgcn_s_wait_asynccnt(N);
#else
  if (N == 0)
    asm volatile("s_wait_asynccnt 0x0" ::: "memory");
  else
    asm volatile("s_wait_asynccnt 0x5" ::: "memory");
#endif
#endif
}

// A-fragment (16x32 bf16, M x K): lane half selects K base {0,8};
// per-lane data = K[base..base+7] ++ K[base+16..base+23]
static __device__ __forceinline__ bf16x16 load_a_frag(const bf16_t* p) {
  bf16x8 lo = *(const bf16x8*)(p);
  bf16x8 hi = *(const bf16x8*)(p + 16);
  return __builtin_shufflevector(lo, hi, 0,1,2,3,4,5,6,7,8,9,10,11,12,13,14,15);
}

static __device__ __forceinline__ f32x8 wmma_bf16(bf16x16 a, bf16x16 b, f32x8 c) {
  return __builtin_amdgcn_wmma_f32_16x16x32_bf16(false, a, false, b, (short)0, c,
                                                 false, false);
}

// ------------------------------ converters ---------------------------------

__global__ void cvt_bf16_kernel(const float4* __restrict__ src,
                                bf16_t* __restrict__ dst, int n4) {
  int i = blockIdx.x * blockDim.x + threadIdx.x;
  if (i >= n4) return;
  float4 v = src[i];
  bf16x4 o = {(bf16_t)v.x, (bf16_t)v.y, (bf16_t)v.z, (bf16_t)v.w};
  *(bf16x4*)(dst + 4 * (size_t)i) = o;
}

__global__ void combine_wo_kernel(const float4* __restrict__ a,
                                  const float4* __restrict__ b,
                                  bf16_t* __restrict__ dst, int n4) {
  int i = blockIdx.x * blockDim.x + threadIdx.x;
  if (i >= n4) return;
  float4 va = a[i];
  float4 vb = b[i];
  bf16x4 o = {(bf16_t)(0.5f * (va.x + vb.x)), (bf16_t)(0.5f * (va.y + vb.y)),
              (bf16_t)(0.5f * (va.z + vb.z)), (bf16_t)(0.5f * (va.w + vb.w))};
  *(bf16x4*)(dst + 4 * (size_t)i) = o;
}

// ------------------------------ RoPE (in place) ----------------------------

__global__ void rope_kernel(bf16_t* __restrict__ Qb, bf16_t* __restrict__ Kb) {
  unsigned idx = blockIdx.x * blockDim.x + threadIdx.x;  // 2^23 threads
  int j   = idx & 63;
  int s   = (idx >> 6) & 2047;
  int bh  = (idx >> 17) & 31;
  int buf = (idx >> 22) & 1;
  bf16_t* p = (buf ? Kb : Qb) + ((size_t)bh * SEQ + s) * HD + j;
  // inv_freq = 10000^(-2j/128)
  float inv_freq = __expf(-(float)(2 * j) * (9.210340371976184f / 128.f));
  float ang = (float)s * inv_freq;
  float sn, cs;
  sincosf(ang, &sn, &cs);
  float x1 = (float)p[0], x2 = (float)p[64];
  p[0]  = (bf16_t)(x1 * cs - x2 * sn);
  p[64] = (bf16_t)(x2 * cs + x1 * sn);
}

// ------------------------------ GEMM ---------------------------------------
// C[M,N] = A[M,K] * B[N,K]^T, K=2048, bf16 in / f32 accum.
// WG tile 64(M) x 256(N), 8 waves arranged 2x4, wave tile 32x64.
// Double-buffered LDS; tile k+1 streamed via async-to-LDS while computing k.
// MODE 0: fp32 out row-major [M][N]
// MODE 1: bf16 out -> [b][h][s][d]     (h = h_off + n>>7, d = n&127)
// MODE 2: bf16 out -> [b][h][d][s]     (V transposed)

template <int MODE>
__global__ __launch_bounds__(256) void gemm_kernel(
    const bf16_t* __restrict__ A, const bf16_t* __restrict__ Bw,
    float* __restrict__ outF, bf16_t* __restrict__ outB,
    int N, int h_off) {
  __shared__ bf16_t As[2][64][40];
  __shared__ bf16_t Bs[2][256][40];
  const int K = HIDDEN;
  int t = threadIdx.x;
  int lane = t & 31, wave = t >> 5;
  int wm = wave >> 2, wn = wave & 3;
  int half = lane >> 4, l16 = lane & 15;
  int kbA = half * 8, kbB = half * 16;
  int m0 = blockIdx.y * 64, n0 = blockIdx.x * 256;

  const f32x8 vz = {};
  f32x8 acc[2][4];
#pragma unroll
  for (int i = 0; i < 2; ++i)
#pragma unroll
    for (int j = 0; j < 4; ++j) acc[i][j] = vz;

  int arow = t >> 2, ach = (t & 3) * 8;
  // 5 x 16B async copies per thread per tile (1 A-chunk + 4 B-chunks)
  auto issue_tile = [&](int k0, int buf) {
    copy16_to_lds(A + (size_t)(m0 + arow) * K + k0 + ach, &As[buf][arow][ach]);
#pragma unroll
    for (int i = 0; i < 4; ++i) {
      int br = arow + i * 64;
      copy16_to_lds(Bw + (size_t)(n0 + br) * K + k0 + ach, &Bs[buf][br][ach]);
    }
  };

  issue_tile(0, 0);
  for (int k0 = 0; k0 < K; k0 += 32) {
    int buf = (k0 >> 5) & 1;
    if (k0 + 32 < K) {
      issue_tile(k0 + 32, buf ^ 1);
      async_wait<5>();  // previous 5 (current tile) complete; next 5 in flight
    } else {
      async_wait<0>();
    }
    __syncthreads();
    bf16x16 af[2], bfr[4];
#pragma unroll
    for (int mi = 0; mi < 2; ++mi)
      af[mi] = load_a_frag(&As[buf][wm * 32 + mi * 16 + l16][kbA]);
#pragma unroll
    for (int ni = 0; ni < 4; ++ni)
      bfr[ni] = load_a_frag(&Bs[buf][wn * 64 + ni * 16 + l16][kbA]);
#pragma unroll
    for (int mi = 0; mi < 2; ++mi)
#pragma unroll
      for (int ni = 0; ni < 4; ++ni)
        acc[mi][ni] = wmma_bf16(af[mi], bfr[ni], acc[mi][ni]);
    __syncthreads();  // all waves done with buf before it is refilled
  }

  // C layout: lane half selects M block of 8, VGPR r = row, lane&15 = col
#pragma unroll
  for (int mi = 0; mi < 2; ++mi) {
#pragma unroll
    for (int ni = 0; ni < 4; ++ni) {
      int mt = m0 + wm * 32 + mi * 16 + half * 8;
      int nt = n0 + wn * 64 + ni * 16 + l16;
#pragma unroll
      for (int r = 0; r < 8; ++r) {
        int m = mt + r;
        float v = acc[mi][ni][r];
        if constexpr (MODE == 0) {
          outF[(size_t)m * N + nt] = v;
        } else if constexpr (MODE == 1) {
          int b = m >> 11, s = m & 2047;
          int h = h_off + (nt >> 7), d = nt & 127;
          outB[((size_t)(b * HEADS + h) * SEQ + s) * HD + d] = (bf16_t)v;
        } else {
          int b = m >> 11, s = m & 2047;
          int h = h_off + (nt >> 7), d = nt & 127;
          outB[((size_t)(b * HEADS + h) * HD + d) * SEQ + s] = (bf16_t)v;
        }
      }
    }
  }
}

// ------------------------------ flash attention ----------------------------
// Block = 4 waves x 32 lanes; each wave owns a 16-row q strip (block = 64 q).
// Grid: (SEQ/64, B*HEADS).

__global__ __launch_bounds__(128) void attn_kernel(
    const bf16_t* __restrict__ Q, const bf16_t* __restrict__ Km,
    const bf16_t* __restrict__ VT, const float* __restrict__ mask,
    bf16_t* __restrict__ O) {
  __shared__ bf16_t Pl[4][16][40];  // per-wave P tile (16 x 32, padded)
  int lane = threadIdx.x & 31, wave = threadIdx.x >> 5;
  int half = lane >> 4, l16 = lane & 15;
  int kbA = half * 8, kbB = half * 16;
  int bh = blockIdx.y, b = bh >> 4, h = bh & 15;
  int q0 = blockIdx.x * 64 + wave * 16;

  const bf16_t* Qp = Q + (size_t)bh * SEQ * HD;
  const bf16_t* Kp = Km + (size_t)bh * SEQ * HD;
  const bf16_t* Vp = VT + (size_t)bh * HD * SEQ;
  const float* Mp = mask + (size_t)b * SEQ * SEQ;

  bf16x16 qf[4];
  {
    const bf16_t* qrow = Qp + (size_t)(q0 + l16) * HD;
#pragma unroll
    for (int ks = 0; ks < 4; ++ks) qf[ks] = load_a_frag(qrow + ks * 32 + kbA);
  }

  const f32x8 vz = {};
  f32x8 acc[8];
#pragma unroll
  for (int dt = 0; dt < 8; ++dt) acc[dt] = vz;
  float mrow[8], lrow[8];
#pragma unroll
  for (int r = 0; r < 8; ++r) { mrow[r] = -1e30f; lrow[r] = 0.f; }
  const float scale = 0.088388347648318447f;  // 128^-0.5

  for (int kj = 0; kj < SEQ; kj += 32) {
    // ---- scores: two 16x16 tiles, contraction over d (4 k-steps of 32) ----
    f32x8 c0 = vz, c1 = vz;
    {
      const bf16_t* kr0 = Kp + (size_t)(kj + l16) * HD + kbA;
      const bf16_t* kr1 = kr0 + (size_t)16 * HD;
#pragma unroll
      for (int ks = 0; ks < 4; ++ks)
        c0 = wmma_bf16(qf[ks], load_a_frag(kr0 + ks * 32), c0);
#pragma unroll
      for (int ks = 0; ks < 4; ++ks)
        c1 = wmma_bf16(qf[ks], load_a_frag(kr1 + ks * 32), c1);
    }
    // ---- online softmax over the 32 new columns ----
    float p0[8], p1[8], alph[8];
#pragma unroll
    for (int r = 0; r < 8; ++r) {
      int qr = q0 + half * 8 + r;
      const float* mp = Mp + (size_t)qr * SEQ + kj;
      float s0 = c0[r] * scale + mp[l16];
      float s1 = c1[r] * scale + mp[16 + l16];
      float mx = fmaxf(s0, s1);
#pragma unroll
      for (int off = 1; off < 16; off <<= 1)
        mx = fmaxf(mx, __shfl_xor(mx, off, 32));
      float mnew = fmaxf(mrow[r], mx);
      float a = __expf(mrow[r] - mnew);
      mrow[r] = mnew;
      float e0 = __expf(s0 - mnew);
      float e1 = __expf(s1 - mnew);
      float sum = e0 + e1;
#pragma unroll
      for (int off = 1; off < 16; off <<= 1) sum += __shfl_xor(sum, off, 32);
      lrow[r] = lrow[r] * a + sum;
      p0[r] = e0; p1[r] = e1; alph[r] = a;
    }
#pragma unroll
    for (int dt = 0; dt < 8; ++dt)
#pragma unroll
      for (int r = 0; r < 8; ++r) acc[dt][r] *= alph[r];

    // ---- P: C-layout -> A-layout via per-wave LDS tile ----
#pragma unroll
    for (int r = 0; r < 8; ++r) {
      int m = half * 8 + r;
      Pl[wave][m][l16] = (bf16_t)p0[r];
      Pl[wave][m][16 + l16] = (bf16_t)p1[r];
    }
    asm volatile("s_wait_dscnt 0" ::: "memory");  // intra-wave LDS RAW
    bf16x16 pf = load_a_frag(&Pl[wave][l16][kbA]);

    // ---- PV: B operand from transposed V (contiguous along s) ----
#pragma unroll
    for (int dt = 0; dt < 8; ++dt) {
      const bf16_t* vr = Vp + (size_t)(dt * 16 + l16) * SEQ + kj + kbA;
      acc[dt] = wmma_bf16(pf, load_a_frag(vr), acc[dt]);
    }
  }

  float inv[8];
#pragma unroll
  for (int r = 0; r < 8; ++r) inv[r] = 1.f / lrow[r];
#pragma unroll
  for (int dt = 0; dt < 8; ++dt) {
#pragma unroll
    for (int r = 0; r < 8; ++r) {
      int s = q0 + half * 8 + r;
      size_t off = (size_t)(b * SEQ + s) * HIDDEN + h * HD + dt * 16 + l16;
      O[off] = (bf16_t)(acc[dt][r] * inv[r]);
    }
  }
}

// ------------------------------ launch -------------------------------------

extern "C" void kernel_launch(void* const* d_in, const int* in_sizes, int n_in,
                              void* d_out, int out_size, void* d_ws,
                              size_t ws_size, hipStream_t stream) {
  (void)in_sizes; (void)n_in; (void)out_size; (void)ws_size;
  const float* x_q  = (const float*)d_in[0];
  const float* x_kv = (const float*)d_in[1];
  const float* mask = (const float*)d_in[2];
  const float* Wq   = (const float*)d_in[3];
  const float* Wks  = (const float*)d_in[4];
  const float* Wvs  = (const float*)d_in[5];
  const float* Wkc  = (const float*)d_in[6];
  const float* Wvc  = (const float*)d_in[7];
  const float* Wos  = (const float*)d_in[8];
  const float* Woc  = (const float*)d_in[9];

  const size_t XE = (size_t)2 * SEQ * HIDDEN;  // 8388608 activations
  const size_t WF = (size_t)HIDDEN * HIDDEN;   // 4194304 full weight
  const size_t WH = WF / 2;                    // half weight (8 heads)

  bf16_t* ws   = (bf16_t*)d_ws;
  bf16_t* xqb  = ws;
  bf16_t* xkb  = xqb + XE;
  bf16_t* wqb  = xkb + XE;
  bf16_t* wksb = wqb + WF;
  bf16_t* wvsb = wksb + WH;
  bf16_t* wkcb = wvsb + WH;
  bf16_t* wvcb = wkcb + WH;
  bf16_t* wob  = wvcb + WH;
  bf16_t* Qb   = wob + WF;
  bf16_t* Kb   = Qb + XE;
  bf16_t* Vt   = Kb + XE;
  bf16_t* Ob   = Vt + XE;

  // 1) conversions (weights: only the surviving half of each K/V projection)
  cvt_bf16_kernel<<<(int)(XE / 4 / 256), 256, 0, stream>>>((const float4*)x_q, xqb, (int)(XE / 4));
  cvt_bf16_kernel<<<(int)(XE / 4 / 256), 256, 0, stream>>>((const float4*)x_kv, xkb, (int)(XE / 4));
  cvt_bf16_kernel<<<(int)(WF / 4 / 256), 256, 0, stream>>>((const float4*)Wq, wqb, (int)(WF / 4));
  cvt_bf16_kernel<<<(int)(WH / 4 / 256), 256, 0, stream>>>((const float4*)Wks, wksb, (int)(WH / 4));
  cvt_bf16_kernel<<<(int)(WH / 4 / 256), 256, 0, stream>>>((const float4*)Wvs, wvsb, (int)(WH / 4));
  cvt_bf16_kernel<<<(int)(WH / 4 / 256), 256, 0, stream>>>((const float4*)Wkc + WH / 4, wkcb, (int)(WH / 4));
  cvt_bf16_kernel<<<(int)(WH / 4 / 256), 256, 0, stream>>>((const float4*)Wvc + WH / 4, wvcb, (int)(WH / 4));
  combine_wo_kernel<<<(int)(WF / 4 / 256), 256, 0, stream>>>((const float4*)Wos, (const float4*)Woc, wob, (int)(WF / 4));

  // 2) projections (M = 4096 rows always)
  gemm_kernel<1><<<dim3(8, 64), 256, 0, stream>>>(xqb, wqb, nullptr, Qb, 2048, 0);
  gemm_kernel<1><<<dim3(4, 64), 256, 0, stream>>>(xqb, wksb, nullptr, Kb, 1024, 0);
  gemm_kernel<1><<<dim3(4, 64), 256, 0, stream>>>(xkb, wkcb, nullptr, Kb, 1024, 8);
  gemm_kernel<2><<<dim3(4, 64), 256, 0, stream>>>(xqb, wvsb, nullptr, Vt, 1024, 0);
  gemm_kernel<2><<<dim3(4, 64), 256, 0, stream>>>(xkb, wvcb, nullptr, Vt, 1024, 8);

  // 3) RoPE in place on Q (all heads) and K (all heads)
  rope_kernel<<<32768, 256, 0, stream>>>(Qb, Kb);

  // 4) flash attention
  attn_kernel<<<dim3(SEQ / 64, 32), 128, 0, stream>>>(Qb, Kb, Vt, mask, Ob);

  // 5) output projection, fp32 result
  gemm_kernel<0><<<dim3(8, 64), 256, 0, stream>>>(Ob, wob, (float*)d_out, nullptr, 2048, 0);
}
